// SpanBERTFullCoref_58110907515618
// MI455X (gfx1250) — matrix operand order repaired and checked by reading, not compiled, not run on hardware
//
#include <hip/hip_runtime.h>
#include <math.h>

// ---------------- types ----------------
typedef __attribute__((ext_vector_type(16))) __bf16 v16bf;
typedef __attribute__((ext_vector_type(8)))  __bf16 v8bf;
typedef __attribute__((ext_vector_type(8)))  float  v8f;

#define LCK   512      // tokens
#define HH    768      // hidden
#define NSP   8192     // spans
#define MM    2324     // span emb dim
#define MP    2432     // padded K for base/c2 (mult of 128)
#define FFD   3000
#define FP    3072     // padded F (12 * 256)
#define KTOP  204
#define KPAD  208
#define CANT  50
#define NPAIR 10200    // 204*50
#define DPAIR 7052     // 3M+80
#define DP    7168     // padded pair K (56*128)

__device__ __forceinline__ unsigned short f2bf(float f) {
  unsigned u = __float_as_uint(f);
  unsigned r = u + 0x7FFFu + ((u >> 16) & 1u);
  return (unsigned short)(r >> 16);
}
__device__ __forceinline__ float bf2f(unsigned short h) {
  return __uint_as_float(((unsigned)h) << 16);
}
__device__ __forceinline__ v8f vzero8() {
  v8f z;
#pragma unroll
  for (int t = 0; t < 8; ++t) z[t] = 0.f;
  return z;
}
// A fragment: two 16-byte chunks (K k..k+7 and k+8..k+15 within lane's half)
__device__ __forceinline__ v16bf load_afrag(const unsigned short* p) {
  v8bf lo = *(const v8bf*)p;
  v8bf hi = *(const v8bf*)(p + 8);
  v16bf a;
#pragma unroll
  for (int t = 0; t < 8; ++t) { a[t] = lo[t]; a[t + 8] = hi[t]; }
  return a;
}

// ---------------- weight convert / transpose (one-shot, bandwidth-trivial) ----
__global__ void k_cvt_mw0(const float* __restrict__ w0, unsigned short* __restrict__ wT) {
  int id = blockIdx.x * 256 + threadIdx.x;
  if (id >= FP * MP) return;
  int f = id / MP, k = id % MP;
  float v = (f < FFD && k < MM) ? w0[(size_t)k * FFD + f] : 0.f;
  wT[id] = f2bf(v);
}
__global__ void k_cvt_c2w(const float* __restrict__ w, unsigned short* __restrict__ wT) {
  int id = blockIdx.x * 256 + threadIdx.x;
  if (id >= MP * MP) return;
  int n = id / MP, k = id % MP;
  float v = (n < MM && k < MM) ? w[(size_t)k * MM + n] : 0.f;
  wT[id] = f2bf(v);
}
__global__ void k_cvt_aw0(const float* __restrict__ w, unsigned short* __restrict__ wT) {
  int id = blockIdx.x * 256 + threadIdx.x;
  if (id >= FP * DP) return;
  int f = id / DP, k = id % DP;
  float v = (f < FFD && k < DPAIR) ? w[(size_t)k * FFD + f] : 0.f;
  wT[id] = f2bf(v);
}

// ---------------- width-prior scalar scores (30 widths) ----------------------
__global__ void k_wprior(const float* __restrict__ wp, const float* __restrict__ ww0,
                         const float* __restrict__ wb0, const float* __restrict__ ww1,
                         const float* __restrict__ wb1, float* __restrict__ wscore) {
  __shared__ float red[256];
  int w = blockIdx.x, tid = threadIdx.x;
  const float* p = wp + w * 20;
  float acc = 0.f;
  for (int f = tid; f < FFD; f += 256) {
    float d = wb0[f];
#pragma unroll
    for (int q = 0; q < 20; ++q) d += p[q] * ww0[q * FFD + f];
    acc += fmaxf(d, 0.f) * ww1[f];
  }
  red[tid] = acc; __syncthreads();
  for (int s = 128; s; s >>= 1) { if (tid < s) red[tid] += red[tid + s]; __syncthreads(); }
  if (tid == 0) wscore[w] = red[0] + wb1[0];
}

// ---------------- span extraction + head attention + base emb (bf16) ---------
__global__ void k_span(const float* __restrict__ h, const int* __restrict__ spans,
                       const float* __restrict__ w_head, const float* __restrict__ b_head,
                       const float* __restrict__ width_emb,
                       unsigned short* __restrict__ base_bf, int* __restrict__ widths) {
  __shared__ float sc_s[32];
  __shared__ float wt_s[32];
  __shared__ float head_s[HH];
  int n = blockIdx.x, tid = threadIdx.x, lane = tid & 31, wv = tid >> 5;
  int s = spans[2 * n]; s = s < 0 ? 0 : (s > LCK - 1 ? LCK - 1 : s);
  int e = spans[2 * n + 1]; e = e < 0 ? 0 : (e > LCK - 1 ? LCK - 1 : e);
  int width = e - s + 1; if (width < 1) width = 1; if (width > 30) width = 30;
  if (tid == 0) widths[n] = width;
  float bh = b_head[0];
  for (int w = wv; w < 30; w += 8) {
    float p = 0.f;
    if (w < width) {
      int row = s + w; if (row > LCK - 1) row = LCK - 1;
      const float* hr = h + row * HH;
      for (int k = lane; k < HH; k += 32) p += hr[k] * w_head[k];
    }
    for (int off = 16; off; off >>= 1) p += __shfl_xor(p, off, 32);
    if (lane == 0) sc_s[w] = p + bh;
  }
  __syncthreads();
  if (tid < 32) {
    float v = (tid < width) ? sc_s[tid] : -1e30f;
    float mx = v;
    for (int off = 16; off; off >>= 1) mx = fmaxf(mx, __shfl_xor(mx, off, 32));
    float ex = (tid < width) ? __expf(v - mx) : 0.f;
    float sm = ex;
    for (int off = 16; off; off >>= 1) sm += __shfl_xor(sm, off, 32);
    wt_s[tid] = ex / (sm + 1e-13f);
  }
  __syncthreads();
  for (int k = tid; k < HH; k += 256) {
    float acc = 0.f;
    for (int w = 0; w < width; ++w) {
      int row = s + w; if (row > LCK - 1) row = LCK - 1;
      acc += wt_s[w] * h[row * HH + k];
    }
    head_s[k] = acc;
  }
  __syncthreads();
  size_t rb = (size_t)n * MP;
  for (int k = tid; k < MP; k += 256) {
    float v;
    if (k < HH)            v = h[s * HH + k];
    else if (k < 2 * HH)   v = h[e * HH + (k - HH)];
    else if (k < 3 * HH)   v = head_s[k - 2 * HH];
    else if (k < MM)       v = width_emb[(width - 1) * 20 + (k - 3 * HH)];
    else                   v = 0.f;
    base_bf[rb + k] = f2bf(v);
  }
}

// ---------------- mention score init + GEMM (WMMA bf16, fused relu.w1) -------
__global__ void k_minit(const int* __restrict__ widths, const float* __restrict__ wscore,
                        const float* __restrict__ mb1, float* __restrict__ m) {
  int n = blockIdx.x * 256 + threadIdx.x;
  if (n < NSP) m[n] = mb1[0] + wscore[widths[n] - 1];
}

__global__ void k_mention(const unsigned short* __restrict__ base_bf,
                          const unsigned short* __restrict__ w0T,
                          const float* __restrict__ b0, const float* __restrict__ w1,
                          float* __restrict__ m) {
  int tid = threadIdx.x, lane = tid & 31, wv = tid >> 5;
  int rowBase = blockIdx.x * 16;
  int col0 = blockIdx.y * 256 + wv * 32;
  v8f acc0 = vzero8(), acc1 = vzero8();
  const unsigned short* aRow = base_bf + (size_t)(rowBase + (lane & 15)) * MP + ((lane >> 4) << 4);
  const unsigned short* bR0  = w0T + (size_t)(col0 + (lane & 15)) * MP + ((lane >> 4) << 4);
  const unsigned short* bR1  = bR0 + (size_t)16 * MP;
  for (int k0 = 0; k0 < MP; k0 += 32) {
    v16bf a  = load_afrag(aRow + k0);
    v16bf b0v = *(const v16bf*)(bR0 + k0);
    v16bf b1v = *(const v16bf*)(bR1 + k0);
    __builtin_prefetch(bR0 + k0 + 256, 0, 1);
    acc0 = __builtin_amdgcn_wmma_f32_16x16x32_bf16(false, a, false, b0v, (short)0, acc0, false, false);
    acc1 = __builtin_amdgcn_wmma_f32_16x16x32_bf16(false, a, false, b1v, (short)0, acc1, false, false);
  }
  int cA = col0 + (lane & 15), cB = cA + 16;
  float w1a = (cA < FFD) ? w1[cA] : 0.f, b0a = (cA < FFD) ? b0[cA] : 0.f;
  float w1b = (cB < FFD) ? w1[cB] : 0.f, b0b = (cB < FFD) ? b0[cB] : 0.f;
#pragma unroll
  for (int r = 0; r < 8; ++r) {
    int row = rowBase + r + ((lane >> 4) << 3);
    float c = fmaxf(acc0[r] + b0a, 0.f) * w1a + fmaxf(acc1[r] + b0b, 0.f) * w1b;
    for (int off = 1; off < 16; off <<= 1) c += __shfl_xor(c, off, 32);
    if ((lane & 15) == 0) atomicAdd(&m[row], c);
  }
}

// ---------------- top-204 selection + document-order compaction --------------
__global__ void k_topk(const float* __restrict__ m, int* __restrict__ top_idx,
                       float* __restrict__ top_m) {
  __shared__ float mw[NSP];
  __shared__ unsigned char flg[NSP];
  __shared__ float rv[1024];
  __shared__ int   ri[1024];
  int tid = threadIdx.x;
  for (int i = tid; i < NSP; i += 1024) { mw[i] = m[i]; flg[i] = 0; }
  __syncthreads();
  for (int k = 0; k < KTOP; ++k) {
    float bv = -3e38f; int bi = 0;
    for (int i = tid; i < NSP; i += 1024) { float v = mw[i]; if (v > bv) { bv = v; bi = i; } }
    rv[tid] = bv; ri[tid] = bi; __syncthreads();
    for (int s = 512; s; s >>= 1) {
      if (tid < s) {
        if (rv[tid + s] > rv[tid] || (rv[tid + s] == rv[tid] && ri[tid + s] < ri[tid])) {
          rv[tid] = rv[tid + s]; ri[tid] = ri[tid + s];
        }
      }
      __syncthreads();
    }
    if (tid == 0) { flg[ri[0]] = 1; mw[ri[0]] = -3e38f; }
    __syncthreads();
  }
  if (tid == 0) {
    int c = 0;
    for (int i = 0; i < NSP && c < KTOP; ++i)
      if (flg[i]) { top_idx[c] = i; top_m[c] = m[i]; ++c; }
    for (; c < KPAD; ++c) { top_idx[c] = 0; top_m[c] = 0.f; }
  }
}

__global__ void k_gather(const int* __restrict__ top_idx, const unsigned short* __restrict__ base_bf,
                         const int* __restrict__ spans, const int* __restrict__ spk_ids,
                         const int* __restrict__ sent_map, unsigned short* __restrict__ tb,
                         int* __restrict__ top_spk, int* __restrict__ top_seg) {
  int k = blockIdx.x, tid = threadIdx.x;
  if (k >= KTOP) {
    for (int q = tid; q < MP; q += 256) tb[(size_t)k * MP + q] = 0;
    if (tid == 0) { top_spk[k] = 0; top_seg[k] = 0; }
    return;
  }
  int idx = top_idx[k];
  for (int q = tid; q < MP; q += 256) tb[(size_t)k * MP + q] = base_bf[(size_t)idx * MP + q];
  if (tid == 0) {
    int s = spans[2 * idx]; s = s < 0 ? 0 : (s > LCK - 1 ? LCK - 1 : s);
    top_spk[k] = spk_ids[s]; top_seg[k] = sent_map[s];
  }
}

// ---------------- coarse: src = top_base @ c2_w + b  (WMMA) ------------------
__global__ void k_coarse(const unsigned short* __restrict__ tb, const unsigned short* __restrict__ c2T,
                         const float* __restrict__ c2b, unsigned short* __restrict__ src_bf) {
  int tid = threadIdx.x, lane = tid & 31, wv = tid >> 5;
  int rowBase = blockIdx.x * 16;
  int col0 = (blockIdx.y * 8 + wv) * 16;
  v8f acc = vzero8();
  const unsigned short* aRow = tb + (size_t)(rowBase + (lane & 15)) * MP + ((lane >> 4) << 4);
  const unsigned short* bRow = c2T + (size_t)(col0 + (lane & 15)) * MP + ((lane >> 4) << 4);
  for (int k0 = 0; k0 < MP; k0 += 32) {
    v16bf a = load_afrag(aRow + k0);
    v16bf b = *(const v16bf*)(bRow + k0);
    acc = __builtin_amdgcn_wmma_f32_16x16x32_bf16(false, a, false, b, (short)0, acc, false, false);
  }
  int col = col0 + (lane & 15);
  float bias = (col < MM) ? c2b[col] : 0.f;
#pragma unroll
  for (int r = 0; r < 8; ++r) {
    int row = rowBase + r + ((lane >> 4) << 3);
    src_bf[(size_t)row * MP + col] = f2bf(acc[r] + bias);
  }
}

// ---------------- cp[i][j] = src.tb^T + m_i + m_j, mask j<i  (WMMA) ----------
__global__ void k_cp(const unsigned short* __restrict__ src_bf, const unsigned short* __restrict__ tb,
                     const float* __restrict__ top_m, float* __restrict__ cp) {
  int tid = threadIdx.x, lane = tid & 31, wv = tid >> 5;
  int tileJ = blockIdx.y * 8 + wv;
  if (tileJ >= KPAD / 16) return;
  int rowBase = blockIdx.x * 16, col0 = tileJ * 16;
  v8f acc = vzero8();
  const unsigned short* aRow = src_bf + (size_t)(rowBase + (lane & 15)) * MP + ((lane >> 4) << 4);
  const unsigned short* bRow = tb + (size_t)(col0 + (lane & 15)) * MP + ((lane >> 4) << 4);
  for (int k0 = 0; k0 < MP; k0 += 32) {
    v16bf a = load_afrag(aRow + k0);
    v16bf b = *(const v16bf*)(bRow + k0);
    acc = __builtin_amdgcn_wmma_f32_16x16x32_bf16(false, a, false, b, (short)0, acc, false, false);
  }
  int j = col0 + (lane & 15);
#pragma unroll
  for (int r = 0; r < 8; ++r) {
    int i = rowBase + r + ((lane >> 4) << 3);
    if (i < KTOP && j < KTOP) {
      float v = acc[r] + top_m[i] + top_m[j];
      if (j >= i) v = -1e9f;
      cp[i * KPAD + j] = v;
    }
  }
}

// ---------------- per-row top-50 antecedents (value-desc, tie low idx) -------
__global__ void k_anttopk(const float* __restrict__ cp, float* __restrict__ ant_sc,
                          int* __restrict__ ant_pos) {
  int i = blockIdx.x, lane = threadIdx.x;
  float lv[7];
#pragma unroll
  for (int q = 0; q < 7; ++q) { int c = lane + q * 32; lv[q] = (c < KTOP) ? cp[i * KPAD + c] : -3e38f; }
  for (int it = 0; it < CANT; ++it) {
    float best = lv[0]; int bslot = 0;
#pragma unroll
    for (int q = 1; q < 7; ++q) if (lv[q] > best) { best = lv[q]; bslot = q; }
    int bcol = lane + bslot * 32;
    for (int off = 16; off; off >>= 1) {
      float ov = __shfl_xor(best, off, 32);
      int oc = __shfl_xor(bcol, off, 32);
      if (ov > best || (ov == best && oc < bcol)) { best = ov; bcol = oc; }
    }
    if (lane == 0) { ant_sc[i * CANT + it] = best; ant_pos[i * CANT + it] = bcol; }
    if (lane == (bcol & 31)) lv[bcol >> 5] = -3e38f;
  }
}

// ---------------- pair metadata + fine init ----------------------------------
__global__ void k_pairmeta(const int* __restrict__ ant_pos, const int* __restrict__ top_spk,
                           const int* __restrict__ top_seg, const float* __restrict__ ab1,
                           int* __restrict__ meta, float* __restrict__ fine) {
  int r = blockIdx.x * 256 + threadIdx.x;
  if (r >= NPAIR) return;
  int i = r / CANT, j = ant_pos[r];
  int same = (top_spk[i] == top_spk[j]) ? 1 : 0;
  int d = i - j; if (d < 0) d = 0;
  int bucket = (d >= 64) ? 9 : (d >= 32) ? 8 : (d >= 16) ? 7 : (d >= 8) ? 6 : (d >= 5) ? 5 : d;
  int sd = top_seg[i] - top_seg[j]; if (sd < 0) sd = 0; if (sd > 2) sd = 2;
  int valid = (j < i) ? 1 : 0;
  meta[r] = (j & 255) | (same << 8) | (bucket << 9) | (sd << 13) | (valid << 15);
  fine[r] = ab1[0];
}

// ---------------- fine pairwise FFNN: on-the-fly A tile in LDS + WMMA --------
__global__ void k_fine(const unsigned short* __restrict__ w0T, const float* __restrict__ b0,
                       const float* __restrict__ w1, const unsigned short* __restrict__ tb,
                       const int* __restrict__ meta,
                       const float* __restrict__ speaker_emb, const float* __restrict__ genre_emb,
                       const float* __restrict__ dist_emb, const float* __restrict__ segment_emb,
                       const int* __restrict__ gen_ids, float* __restrict__ fine) {
  __shared__ unsigned short As[16][128];
  __shared__ int rowI[16], rowJ[16];
  __shared__ float featS[16][80];
  int tid = threadIdx.x, lane = tid & 31, wv = tid >> 5;
  int pairBase = blockIdx.x * 16;
  int col0 = blockIdx.y * 256 + wv * 32;
  int g = gen_ids[0]; if (g < 0) g = 0; if (g > 6) g = 6;
  if (tid < 16) {
    int pr = pairBase + tid;
    int mw = (pr < NPAIR) ? meta[pr] : 0;
    rowJ[tid] = mw & 255;
    rowI[tid] = (pr < NPAIR) ? (pr / CANT) : 0;
  }
  __syncthreads();
  for (int q = tid; q < 16 * 80; q += 256) {
    int rr = q / 80, f = q % 80;
    int pr = pairBase + rr;
    int mw = (pr < NPAIR) ? meta[pr] : 0;
    int same = (mw >> 8) & 1, bucket = (mw >> 9) & 15, sd = (mw >> 13) & 3;
    float v;
    if (f < 20)      v = speaker_emb[same * 20 + f];
    else if (f < 40) v = genre_emb[g * 20 + (f - 20)];
    else if (f < 60) v = dist_emb[bucket * 20 + (f - 40)];
    else             v = segment_emb[sd * 20 + (f - 60)];
    featS[rr][f] = v;
  }
  __syncthreads();
  v8f acc0 = vzero8(), acc1 = vzero8();
  const unsigned short* bR0 = w0T + (size_t)(col0 + (lane & 15)) * DP + ((lane >> 4) << 4);
  const unsigned short* bR1 = bR0 + (size_t)16 * DP;
  for (int st = 0; st < DP / 128; ++st) {
    int k0 = st * 128;
    for (int q = tid; q < 2048; q += 256) {
      int rr = q >> 7, kk = q & 127;
      int d = k0 + kk;
      unsigned short val;
      if (d < MM)            val = tb[(size_t)rowI[rr] * MP + d];
      else if (d < 2 * MM)   val = tb[(size_t)rowJ[rr] * MP + (d - MM)];
      else if (d < 3 * MM) {
        int dd = d - 2 * MM;
        float gi = bf2f(tb[(size_t)rowI[rr] * MP + dd]);
        float gj = bf2f(tb[(size_t)rowJ[rr] * MP + dd]);
        val = f2bf(gi * gj);
      } else if (d < DPAIR)  val = f2bf(featS[rr][d - 3 * MM]);
      else                   val = 0;
      As[rr][kk] = val;
    }
    __syncthreads();
#pragma unroll
    for (int ks = 0; ks < 4; ++ks) {
      const unsigned short* pa = &As[lane & 15][ks * 32 + ((lane >> 4) << 4)];
      v16bf a = load_afrag(pa);
      v16bf b0v = *(const v16bf*)(bR0 + k0 + ks * 32);
      v16bf b1v = *(const v16bf*)(bR1 + k0 + ks * 32);
      __builtin_prefetch(bR0 + k0 + ks * 32 + 512, 0, 1);
      acc0 = __builtin_amdgcn_wmma_f32_16x16x32_bf16(false, a, false, b0v, (short)0, acc0, false, false);
      acc1 = __builtin_amdgcn_wmma_f32_16x16x32_bf16(false, a, false, b1v, (short)0, acc1, false, false);
    }
    __syncthreads();
  }
  int cA = col0 + (lane & 15), cB = cA + 16;
  float w1a = (cA < FFD) ? w1[cA] : 0.f, b0a = (cA < FFD) ? b0[cA] : 0.f;
  float w1b = (cB < FFD) ? w1[cB] : 0.f, b0b = (cB < FFD) ? b0[cB] : 0.f;
#pragma unroll
  for (int r = 0; r < 8; ++r) {
    int pr = pairBase + r + ((lane >> 4) << 3);
    float c = fmaxf(acc0[r] + b0a, 0.f) * w1a + fmaxf(acc1[r] + b0b, 0.f) * w1b;
    for (int off = 1; off < 16; off <<= 1) c += __shfl_xor(c, off, 32);
    if ((lane & 15) == 0 && pr < NPAIR) atomicAdd(&fine[pr], c);
  }
}

// ---------------- final assembly ---------------------------------------------
__global__ void k_final(const int* __restrict__ meta, const float* __restrict__ ant_sc,
                        const float* __restrict__ fine, const float* __restrict__ dummy,
                        float* __restrict__ out) {
  int id = blockIdx.x * 256 + threadIdx.x;
  if (id >= KTOP * (CANT + 1)) return;
  int i = id / (CANT + 1), c = id % (CANT + 1);
  if (c == 0) { out[id] = dummy[0]; return; }
  int r = i * CANT + (c - 1);
  int valid = (meta[r] >> 15) & 1;
  out[id] = valid ? (ant_sc[r] + fine[r] * 2.0f) : -1e9f;  // /ANT_TEMP = *2
}

// ---------------- host launch ------------------------------------------------
extern "C" void kernel_launch(void* const* d_in, const int* in_sizes, int n_in,
                              void* d_out, int out_size, void* d_ws, size_t ws_size,
                              hipStream_t stream) {
  const float* h          = (const float*)d_in[0];
  const int*   spans      = (const int*)d_in[1];
  const int*   spk_ids    = (const int*)d_in[2];
  const int*   gen_ids    = (const int*)d_in[3];
  const int*   sent_map   = (const int*)d_in[4];
  const float* w_head     = (const float*)d_in[5];
  const float* b_head     = (const float*)d_in[6];
  const float* width_emb  = (const float*)d_in[7];
  const float* width_prior= (const float*)d_in[8];
  const float* speaker_emb= (const float*)d_in[9];
  const float* segment_emb= (const float*)d_in[10];
  const float* genre_emb  = (const float*)d_in[11];
  const float* dist_emb   = (const float*)d_in[12];
  const float* mention_w0 = (const float*)d_in[13];
  const float* mention_b0 = (const float*)d_in[14];
  const float* mention_w1 = (const float*)d_in[15];
  const float* mention_b1 = (const float*)d_in[16];
  const float* width_w0   = (const float*)d_in[17];
  const float* width_b0   = (const float*)d_in[18];
  const float* width_w1   = (const float*)d_in[19];
  const float* width_b1   = (const float*)d_in[20];
  const float* c2_w       = (const float*)d_in[21];
  const float* c2_b       = (const float*)d_in[22];
  const float* ant_w0     = (const float*)d_in[23];
  const float* ant_b0     = (const float*)d_in[24];
  const float* ant_w1     = (const float*)d_in[25];
  const float* ant_b1     = (const float*)d_in[26];
  const float* dummy_bias = (const float*)d_in[27];
  float* out = (float*)d_out;

  size_t off = 0;
  char* base = (char*)d_ws;
  auto alloc = [&](size_t bytes) -> void* {
    void* p = base + off; off += (bytes + 255) & ~(size_t)255; return p;
  };
  unsigned short* base_bf = (unsigned short*)alloc((size_t)NSP * MP * 2);
  unsigned short* mw0T    = (unsigned short*)alloc((size_t)FP * MP * 2);
  unsigned short* c2T     = (unsigned short*)alloc((size_t)MP * MP * 2);
  unsigned short* aw0T    = (unsigned short*)alloc((size_t)FP * DP * 2);
  int*   widths  = (int*)alloc(NSP * 4);
  float* m       = (float*)alloc(NSP * 4);
  float* wscore  = (float*)alloc(32 * 4);
  int*   top_idx = (int*)alloc(KPAD * 4);
  float* top_m   = (float*)alloc(KPAD * 4);
  int*   top_spk = (int*)alloc(KPAD * 4);
  int*   top_seg = (int*)alloc(KPAD * 4);
  unsigned short* tb     = (unsigned short*)alloc((size_t)KPAD * MP * 2);
  unsigned short* src_bf = (unsigned short*)alloc((size_t)KPAD * MP * 2);
  float* cp      = (float*)alloc(KPAD * KPAD * 4);
  float* ant_sc  = (float*)alloc(NPAIR * 4);
  int*   ant_pos = (int*)alloc(NPAIR * 4);
  int*   meta    = (int*)alloc(NPAIR * 4);
  float* fine    = (float*)alloc(NPAIR * 4);

  // 1) weight conversions (bf16, K-major)
  k_cvt_mw0<<<(FP * MP + 255) / 256, 256, 0, stream>>>(mention_w0, mw0T);
  k_cvt_c2w<<<(MP * MP + 255) / 256, 256, 0, stream>>>(c2_w, c2T);
  k_cvt_aw0<<<(FP * DP + 255) / 256, 256, 0, stream>>>(ant_w0, aw0T);
  // 2) width-prior scalar scores
  k_wprior<<<30, 256, 0, stream>>>(width_prior, width_w0, width_b0, width_w1, width_b1, wscore);
  // 3) span embeddings
  k_span<<<NSP, 256, 0, stream>>>(h, spans, w_head, b_head, width_emb, base_bf, widths);
  // 4) mention scores (init + fused WMMA FFNN)
  k_minit<<<NSP / 256, 256, 0, stream>>>(widths, wscore, mention_b1, m);
  k_mention<<<dim3(NSP / 16, FP / 256), 256, 0, stream>>>(base_bf, mw0T, mention_b0, mention_w1, m);
  // 5) top-204 pruning (document order)
  k_topk<<<1, 1024, 0, stream>>>(m, top_idx, top_m);
  k_gather<<<KPAD, 256, 0, stream>>>(top_idx, base_bf, spans, spk_ids, sent_map, tb, top_spk, top_seg);
  // 6) coarse bilinear scores
  k_coarse<<<dim3(KPAD / 16, (MP / 16 + 7) / 8), 256, 0, stream>>>(tb, c2T, c2_b, src_bf);
  k_cp<<<dim3(KPAD / 16, 2), 256, 0, stream>>>(src_bf, tb, top_m, cp);
  // 7) top-50 antecedents per span
  k_anttopk<<<KTOP, 32, 0, stream>>>(cp, ant_sc, ant_pos);
  // 8) pair metadata + fine FFNN
  k_pairmeta<<<(NPAIR + 255) / 256, 256, 0, stream>>>(ant_pos, top_spk, top_seg, ant_b1, meta, fine);
  k_fine<<<dim3((NPAIR + 15) / 16, FP / 256), 256, 0, stream>>>(aw0T, ant_b0, ant_w1, tb, meta,
      speaker_emb, genre_emb, dist_emb, segment_emb, gen_ids, fine);
  // 9) final output [204, 51]
  k_final<<<(KTOP * (CANT + 1) + 255) / 256, 256, 0, stream>>>(meta, ant_sc, fine, dummy_bias, out);
}
